// DecoderRNN_25400436588756
// MI455X (gfx1250) — matrix operand order repaired and checked by reading, not compiled, not run on hardware
//
#include <hip/hip_runtime.h>
#include <cstdint>

#define H512 512
#define B64  64
#define S400 400
#define V50K 50000

typedef _Float16 v16h __attribute__((ext_vector_type(16)));
typedef float    v8f  __attribute__((ext_vector_type(8)));
typedef unsigned int u32x4 __attribute__((ext_vector_type(4)));

// ---------------------------------------------------------------------------
// K1: single-step GRU cell (PyTorch gate order r,z,n), exact fp32.
// grid=64 blocks (8 h-values each, all 64 batches), block=256 threads.
// embedded/h0 chunks staged in LDS (stride 65 to dodge bank conflicts);
// weight reads are uniform across the 64 b-lanes -> broadcast.
// ---------------------------------------------------------------------------
__global__ void gru_kernel(const float* __restrict__ embedded,
                           const float* __restrict__ hidden,
                           const float* __restrict__ w_ih,
                           const float* __restrict__ w_hh,
                           const float* __restrict__ b_ih,
                           const float* __restrict__ b_hh,
                           float* __restrict__ h_new_ws,
                           float* __restrict__ h_new_out) {
  __shared__ float emb_s[64][65];
  __shared__ float h0_s[64][65];
  const int t = threadIdx.x;
  const int hbase = blockIdx.x * 8;

  float acc[2][6];
#pragma unroll
  for (int u = 0; u < 2; ++u)
#pragma unroll
    for (int g = 0; g < 6; ++g) acc[u][g] = 0.0f;

  for (int kc = 0; kc < H512; kc += 64) {
    for (int j = t; j < 64 * 64; j += 256) {
      int bb = j >> 6, ii = j & 63;
      emb_s[bb][ii] = embedded[bb * H512 + kc + ii];
      h0_s[bb][ii]  = hidden[bb * H512 + kc + ii];
    }
    __syncthreads();
#pragma unroll
    for (int u = 0; u < 2; ++u) {
      const int o = t + u * 256;
      const int b = o & 63;
      const int h = hbase + (o >> 6);
      const float* wi0 = w_ih + (size_t)(0 * H512 + h) * H512 + kc;
      const float* wi1 = w_ih + (size_t)(1 * H512 + h) * H512 + kc;
      const float* wi2 = w_ih + (size_t)(2 * H512 + h) * H512 + kc;
      const float* wh0 = w_hh + (size_t)(0 * H512 + h) * H512 + kc;
      const float* wh1 = w_hh + (size_t)(1 * H512 + h) * H512 + kc;
      const float* wh2 = w_hh + (size_t)(2 * H512 + h) * H512 + kc;
      for (int i = 0; i < 64; ++i) {
        float e  = emb_s[b][i];
        float hv = h0_s[b][i];
        acc[u][0] += e  * wi0[i];
        acc[u][1] += e  * wi1[i];
        acc[u][2] += e  * wi2[i];
        acc[u][3] += hv * wh0[i];
        acc[u][4] += hv * wh1[i];
        acc[u][5] += hv * wh2[i];
      }
    }
    __syncthreads();
  }

#pragma unroll
  for (int u = 0; u < 2; ++u) {
    const int o = t + u * 256;
    const int b = o & 63;
    const int h = hbase + (o >> 6);
    float gr = (acc[u][0] + b_ih[h])            + (acc[u][3] + b_hh[h]);
    float gz = (acc[u][1] + b_ih[H512 + h])     + (acc[u][4] + b_hh[H512 + h]);
    float r  = 1.0f / (1.0f + expf(-gr));
    float z  = 1.0f / (1.0f + expf(-gz));
    float n  = tanhf((acc[u][2] + b_ih[2 * H512 + h]) +
                     r * (acc[u][5] + b_hh[2 * H512 + h]));
    float h0v = hidden[b * H512 + h];
    float hn  = (1.0f - z) * n + z * h0v;
    h_new_ws[b * H512 + h]  = hn;
    h_new_out[b * H512 + h] = hn;
  }
}

// ---------------------------------------------------------------------------
// K2: tmp = h_new @ bilin_w   (64x512 @ 512x512), exact fp32.
// ---------------------------------------------------------------------------
__global__ void bilin_tmp_kernel(const float* __restrict__ h_new,
                                 const float* __restrict__ bilin_w,
                                 float* __restrict__ tmp) {
  __shared__ float hn_s[64][65];
  const int t = threadIdx.x;
  const int hbase = blockIdx.x * 8;
  float acc[2] = {0.0f, 0.0f};

  for (int kc = 0; kc < H512; kc += 64) {
    for (int j = t; j < 64 * 64; j += 256) {
      int bb = j >> 6, ii = j & 63;
      hn_s[bb][ii] = h_new[bb * H512 + kc + ii];
    }
    __syncthreads();
#pragma unroll
    for (int u = 0; u < 2; ++u) {
      const int o = t + u * 256;
      const int b = o & 63;
      const int h = hbase + (o >> 6);
      const float* w = bilin_w + (size_t)kc * H512 + h;
      for (int i = 0; i < 64; ++i)
        acc[u] += hn_s[b][i] * w[(size_t)i * H512];
    }
    __syncthreads();
  }
#pragma unroll
  for (int u = 0; u < 2; ++u) {
    const int o = t + u * 256;
    const int b = o & 63;
    const int h = hbase + (o >> 6);
    tmp[b * H512 + h] = acc[u];
  }
}

// ---------------------------------------------------------------------------
// K3: energy[s,b] = tmp[b,:] . enc[s,b,:] + bias. One wave32 per (s,b).
// ---------------------------------------------------------------------------
__global__ void energy_kernel(const float* __restrict__ tmp,
                              const float* __restrict__ enc,
                              const float* __restrict__ bilin_b,
                              float* __restrict__ energy) {
  const int wid  = blockIdx.x * 8 + (threadIdx.x >> 5);
  const int lane = threadIdx.x & 31;
  const int s = wid >> 6;
  const int b = wid & 63;
  const float* tr = tmp + (size_t)b * H512;
  const float* er = enc + (size_t)(s * B64 + b) * H512;
  float sum = 0.0f;
  for (int h = lane; h < H512; h += 32) sum += tr[h] * er[h];
#pragma unroll
  for (int m = 16; m > 0; m >>= 1) sum += __shfl_xor(sum, m, 32);
  if (lane == 0) energy[s * B64 + b] = sum + bilin_b[0];
}

// ---------------------------------------------------------------------------
// K4: softmax over S per batch column. grid=64, block=128.
// ---------------------------------------------------------------------------
__global__ void softmax_kernel(const float* __restrict__ energy,
                               float* __restrict__ attn_ws,
                               float* __restrict__ attn_out) {
  __shared__ float red[128];
  const int b = blockIdx.x, t = threadIdx.x;
  float m = -3.4e38f;
  for (int s = t; s < S400; s += 128) m = fmaxf(m, energy[s * B64 + b]);
  red[t] = m; __syncthreads();
  for (int o = 64; o > 0; o >>= 1) {
    if (t < o) red[t] = fmaxf(red[t], red[t + o]);
    __syncthreads();
  }
  m = red[0]; __syncthreads();
  float sum = 0.0f;
  for (int s = t; s < S400; s += 128) sum += expf(energy[s * B64 + b] - m);
  red[t] = sum; __syncthreads();
  for (int o = 64; o > 0; o >>= 1) {
    if (t < o) red[t] += red[t + o];
    __syncthreads();
  }
  const float inv = 1.0f / red[0];
  for (int s = t; s < S400; s += 128) {
    float a = expf(energy[s * B64 + b] - m) * inv;
    attn_ws[s * B64 + b]  = a;
    attn_out[s * B64 + b] = a;
  }
}

// ---------------------------------------------------------------------------
// K5: context[b,h] = sum_s attn[s,b]*enc[s,b,h]; pack combined = [h_new|ctx]
// as f16 hi/lo split (x = hi + lo) for the WMMA GEMM. grid=64, block=256.
// enc is L2-resident (52 MB < 192 MB) from K3's pass.
// ---------------------------------------------------------------------------
__global__ void context_pack_kernel(const float* __restrict__ attn,
                                    const float* __restrict__ enc,
                                    const float* __restrict__ h_new,
                                    _Float16* __restrict__ chi,
                                    _Float16* __restrict__ clo) {
  __shared__ float a_s[S400];
  const int b = blockIdx.x, t = threadIdx.x;
  for (int s = t; s < S400; s += 256) a_s[s] = attn[s * B64 + b];
  __syncthreads();
  const int h0i = t, h1i = t + 256;
  float acc0 = 0.0f, acc1 = 0.0f;
  for (int s = 0; s < S400; ++s) {
    const float* er = enc + (size_t)(s * B64 + b) * H512;
    float a = a_s[s];
    acc0 += a * er[h0i];
    acc1 += a * er[h1i];
  }
  const size_t base = (size_t)b * 1024;
#pragma unroll
  for (int u = 0; u < 2; ++u) {
    const int h = (u == 0) ? h0i : h1i;
    float x = h_new[b * H512 + h];
    _Float16 hi = (_Float16)x;
    chi[base + h] = hi;
    clo[base + h] = (_Float16)(x - (float)hi);
    float c = (u == 0) ? acc0 : acc1;
    _Float16 chi2 = (_Float16)c;
    chi[base + H512 + h] = chi2;
    clo[base + H512 + h] = (_Float16)(c - (float)chi2);
  }
}

// ---------------------------------------------------------------------------
// K6: logits = combined @ out_w.T + out_b via v_wmma_f32_16x16x32_f16 with
// hi/lo split (3 WMMA per K-step). grid = 50000/16 = 3125 blocks,
// block = 128 threads = 4 waves (one 16-row M-tile each; full M=64).
// B tile (out_w) staged + split in LDS once per block, shared by all waves.
// ---------------------------------------------------------------------------
__global__ void out_gemm_kernel(const _Float16* __restrict__ chi,
                                const _Float16* __restrict__ clo,
                                const float* __restrict__ out_w,
                                const float* __restrict__ out_b,
                                float* __restrict__ logits) {
  __shared__ __align__(16) _Float16 Bhi[16][32];
  __shared__ __align__(16) _Float16 Blo[16][32];
  const int t    = threadIdx.x;
  const int lane = t & 31;
  const int wave = t >> 5;          // 0..3
  const int n0   = blockIdx.x * 16; // output column tile
  const int m0   = wave * 16;       // output row tile
  const int half = lane >> 4;       // lane group (ISA 16-bit A/B layout)
  const int nn   = lane & 15;
  const int bn   = t >> 3;          // B-load: column 0..15
  const int bk   = (t & 7) * 4;     // B-load: k offset 0..28

  v8f acc = {};

  for (int k0 = 0; k0 < 1024; k0 += 32) {
    // --- stage + hi/lo-split B tile (16 cols x 32 K, fp32 -> f16+f16) ---
    const float* src = out_w + (size_t)(n0 + bn) * 1024 + k0 + bk;
    float4 w4 = *(const float4*)src;
    __builtin_prefetch(src + 32, 0, 3);   // next K tile -> global_prefetch
    {
      _Float16 h;
      h = (_Float16)w4.x; Bhi[bn][bk + 0] = h; Blo[bn][bk + 0] = (_Float16)(w4.x - (float)h);
      h = (_Float16)w4.y; Bhi[bn][bk + 1] = h; Blo[bn][bk + 1] = (_Float16)(w4.y - (float)h);
      h = (_Float16)w4.z; Bhi[bn][bk + 2] = h; Blo[bn][bk + 2] = (_Float16)(w4.z - (float)h);
      h = (_Float16)w4.w; Bhi[bn][bk + 3] = h; Blo[bn][bk + 3] = (_Float16)(w4.w - (float)h);
    }
    __syncthreads();

    // --- A fragments: ISA 16-bit A layout => elems[0..7] at K=k0+half*8,
    //     elems[8..15] at K=k0+16+half*8 (two contiguous 16B loads) ---
    union { v16h v; u32x4 q[2]; } afh, afl, bfh, bfl;
    const _Float16* ah = chi + (size_t)(m0 + nn) * 1024 + k0 + half * 8;
    const _Float16* al = clo + (size_t)(m0 + nn) * 1024 + k0 + half * 8;
    afh.q[0] = *(const u32x4*)(ah);
    afh.q[1] = *(const u32x4*)(ah + 16);
    afl.q[0] = *(const u32x4*)(al);
    afl.q[1] = *(const u32x4*)(al + 16);

    // --- B fragments: same K order as A => elems[0..7] at K=half*8, elems[8..15] at K=16+half*8 ---
    const _Float16* bh = &Bhi[nn][half * 8];
    const _Float16* bl = &Blo[nn][half * 8];
    bfh.q[0] = *(const u32x4*)(bh);
    bfh.q[1] = *(const u32x4*)(bh + 16);
    bfl.q[0] = *(const u32x4*)(bl);
    bfl.q[1] = *(const u32x4*)(bl + 16);

    // hi*hi + hi*lo + lo*hi  ~= fp32 product, fp32 accumulate
    acc = __builtin_amdgcn_wmma_f32_16x16x32_f16(false, afh.v, false, bfh.v,
                                                 (short)0, acc, false, false);
    acc = __builtin_amdgcn_wmma_f32_16x16x32_f16(false, afh.v, false, bfl.v,
                                                 (short)0, acc, false, false);
    acc = __builtin_amdgcn_wmma_f32_16x16x32_f16(false, afl.v, false, bfh.v,
                                                 (short)0, acc, false, false);
    __syncthreads();
  }

  // D layout: lanes 0-15 hold M=r, lanes 16-31 hold M=r+8, N = lane&15
  const int col = n0 + nn;
  const float bias = out_b[col];
#pragma unroll
  for (int r = 0; r < 8; ++r) {
    int row = m0 + r + half * 8;
    logits[(size_t)row * V50K + col] = acc[r] + bias;
  }
}

// ---------------------------------------------------------------------------
// K7: in-place log_softmax over V per row. grid=64, block=256.
// ---------------------------------------------------------------------------
__global__ void logsoftmax_kernel(float* __restrict__ logits) {
  __shared__ float red[256];
  const int b = blockIdx.x, t = threadIdx.x;
  float* row = logits + (size_t)b * V50K;
  float m = -3.4e38f;
  for (int v = t; v < V50K; v += 256) m = fmaxf(m, row[v]);
  red[t] = m; __syncthreads();
  for (int o = 128; o > 0; o >>= 1) {
    if (t < o) red[t] = fmaxf(red[t], red[t + o]);
    __syncthreads();
  }
  m = red[0]; __syncthreads();
  float s = 0.0f;
  for (int v = t; v < V50K; v += 256) s += expf(row[v] - m);
  red[t] = s; __syncthreads();
  for (int o = 128; o > 0; o >>= 1) {
    if (t < o) red[t] += red[t + o];
    __syncthreads();
  }
  const float lse = m + logf(red[0]);
  for (int v = t; v < V50K; v += 256) row[v] -= lse;
}

// ---------------------------------------------------------------------------
extern "C" void kernel_launch(void* const* d_in, const int* in_sizes, int n_in,
                              void* d_out, int out_size, void* d_ws, size_t ws_size,
                              hipStream_t stream) {
  const float* embedded = (const float*)d_in[0];
  const float* hidden   = (const float*)d_in[1];
  const float* enc      = (const float*)d_in[2];
  const float* w_ih     = (const float*)d_in[3];
  const float* w_hh     = (const float*)d_in[4];
  const float* b_ih     = (const float*)d_in[5];
  const float* b_hh     = (const float*)d_in[6];
  const float* bilin_w  = (const float*)d_in[7];
  const float* bilin_b  = (const float*)d_in[8];
  const float* out_w    = (const float*)d_in[9];
  const float* out_b    = (const float*)d_in[10];

  float* out      = (float*)d_out;
  float* logp     = out;                       // [64, 50000]
  float* hnew_out = out + (size_t)B64 * V50K;  // [1, 64, 512]
  float* attn_out = hnew_out + B64 * H512;     // [400, 64, 1]

  char* ws = (char*)d_ws;
  float*    h_new  = (float*)(ws + 0);       // 131072 B
  float*    tmp    = (float*)(ws + 131072);  // 131072 B
  float*    energy = (float*)(ws + 262144);  // 102400 B
  float*    attn   = (float*)(ws + 364544);  // 102400 B
  _Float16* chi    = (_Float16*)(ws + 466944); // 131072 B
  _Float16* clo    = (_Float16*)(ws + 598016); // 131072 B

  gru_kernel<<<64, 256, 0, stream>>>(embedded, hidden, w_ih, w_hh, b_ih, b_hh,
                                     h_new, hnew_out);
  bilin_tmp_kernel<<<64, 256, 0, stream>>>(h_new, bilin_w, tmp);
  energy_kernel<<<(S400 * B64) / 8, 256, 0, stream>>>(tmp, enc, bilin_b, energy);
  softmax_kernel<<<64, 128, 0, stream>>>(energy, attn, attn_out);
  context_pack_kernel<<<64, 256, 0, stream>>>(attn, enc, h_new, chi, clo);
  out_gemm_kernel<<<V50K / 16, 128, 0, stream>>>(chi, clo, out_w, out_b, logp);
  logsoftmax_kernel<<<64, 256, 0, stream>>>(logp);
}
